// PointNet2_cls_22007412425308
// MI455X (gfx1250) — matrix operand order repaired and checked by reading, not compiled, not run on hardware
//
#include <hip/hip_runtime.h>
#include <cstdint>
#include <cstddef>

// ---------------------------------------------------------------------------
// PointNet++ classification forward for MI455X (gfx1250, wave32, WMMA).
// All shared-MLP / FC matmuls use v_wmma_f32_16x16x32_f16 with f32 accum.
// Each wave computes a 16x64 C strip: one A fragment feeds 4 independent
// WMMA accumulator chains (4x A reuse), fully unrolled / branch-free so the
// scheduler can clause loads and fill WMMA hazard slots.
// All Opad are multiples of 64 (fc3 padded 12 -> 64).
// ---------------------------------------------------------------------------

typedef __attribute__((ext_vector_type(16))) _Float16 v16h;
typedef __attribute__((ext_vector_type(8)))  float    v8f;
typedef __attribute__((ext_vector_type(8)))  uint32_t v8u;

// ----------------------------- problem constants ---------------------------
static constexpr int BATCH = 32;
static constexpr int NPTS  = 4096;
static constexpr int S1 = 512, K1 = 32;
static constexpr int S2 = 128, K2 = 64;
static constexpr int NS3 = 128;              // points entering SA3 (group-all)

// ----------------------------- workspace layout ----------------------------
static constexpr size_t al256(size_t x) { return (x + 255) & ~size_t(255); }

static constexpr size_t O_PTS  = 0;                                           // [B,N,3] f32
static constexpr size_t O_FID1 = al256(O_PTS  + sizeof(float) * BATCH * NPTS * 3);
static constexpr size_t O_FID2 = al256(O_FID1 + sizeof(int) * BATCH * S1);
static constexpr size_t O_NX1  = al256(O_FID2 + sizeof(int) * BATCH * S2);
static constexpr size_t O_NX2  = al256(O_NX1  + sizeof(float) * BATCH * S1 * 3);
static constexpr size_t O_BQ1  = al256(O_NX2  + sizeof(float) * BATCH * S2 * 3);
static constexpr size_t O_BQ2  = al256(O_BQ1  + sizeof(int) * BATCH * S1 * K1);
static constexpr size_t O_L1P  = al256(O_BQ2  + sizeof(int) * BATCH * S2 * K2);
static constexpr size_t O_L2P  = al256(O_L1P  + sizeof(float) * BATCH * S1 * 128);
static constexpr size_t O_STAT = al256(O_L2P  + sizeof(float) * BATCH * S2 * 256);
static constexpr size_t O_WH   = al256(O_STAT + sizeof(float) * BATCH * 1024 * 2);
static constexpr size_t O_BH   = al256(O_WH   + sizeof(_Float16) * 1024 * 1024);
static constexpr size_t O_BUFA = al256(O_BH   + sizeof(float) * 1024);
static constexpr size_t HALF_BUF = 67108864;  // max M*Cpad halves per stage
static constexpr size_t O_BUFB = al256(O_BUFA + sizeof(_Float16) * HALF_BUF);

// ----------------------------- kernels -------------------------------------

// xyz [B,3,N] -> pts [B,N,3]
__global__ void transpose_xyz_kernel(const float* __restrict__ xyz,
                                     float* __restrict__ pts) {
  int idx = blockIdx.x * blockDim.x + threadIdx.x;
  if (idx >= BATCH * NPTS) return;
  int b = idx / NPTS, n = idx % NPTS;
  float* dst = pts + (size_t)idx * 3;
  const float* src = xyz + (size_t)b * 3 * NPTS + n;
  dst[0] = src[0];
  dst[1] = src[NPTS];
  dst[2] = src[2 * NPTS];
}

// Farthest point sampling: one block per batch; running min-distance array in
// LDS; per-round argmax via wave32 shuffles (2 barriers/round).
__global__ void fps_kernel(const float* __restrict__ pts,
                           int* __restrict__ idx_out, int N, int npoint) {
  __shared__ float dist[NPTS];
  __shared__ float rv[8];
  __shared__ int   ri[8];
  __shared__ int   far_sh;
  const int b = blockIdx.x, tid = threadIdx.x;
  const int lane = tid & 31, wv = tid >> 5;
  const float* P = pts + (size_t)b * N * 3;
  for (int j = tid; j < N; j += 256) dist[j] = 1.0e10f;
  __syncthreads();
  int far = 0;
  for (int it = 0; it < npoint; ++it) {
    if (tid == 0) idx_out[b * npoint + it] = far;   // scan stores carry-in
    const float cx = P[far * 3 + 0], cy = P[far * 3 + 1], cz = P[far * 3 + 2];
    float best = -1.0f; int bi = 0x7fffffff;
    for (int j = tid; j < N; j += 256) {
      float dx = P[j * 3 + 0] - cx, dy = P[j * 3 + 1] - cy, dz = P[j * 3 + 2] - cz;
      float d = dx * dx + dy * dy + dz * dz;
      float nd = fminf(dist[j], d);
      dist[j] = nd;
      if (nd > best) { best = nd; bi = j; }
    }
    // intra-wave argmax (first-max tie break = smaller index)
#pragma unroll
    for (int off = 16; off > 0; off >>= 1) {
      float ov = __shfl_down(best, off);
      int   oi = __shfl_down(bi, off);
      if (ov > best || (ov == best && oi < bi)) { best = ov; bi = oi; }
    }
    if (lane == 0) { rv[wv] = best; ri[wv] = bi; }
    __syncthreads();
    if (tid == 0) {
      float bb = rv[0]; int ii = ri[0];
#pragma unroll
      for (int w = 1; w < 8; ++w) {
        if (rv[w] > bb || (rv[w] == bb && ri[w] < ii)) { bb = rv[w]; ii = ri[w]; }
      }
      far_sh = ii;
    }
    __syncthreads();
    far = far_sh;
  }
}

// gather centroids: nxyz[b,s,:] = pts[b, fidx[b,s], :]
__global__ void gather3_kernel(const float* __restrict__ pts,
                               const int* __restrict__ fidx,
                               float* __restrict__ nxyz, int N, int S) {
  int idx = blockIdx.x * blockDim.x + threadIdx.x;
  if (idx >= BATCH * S) return;
  int b = idx / S;
  int j = fidx[idx];
  const float* src = pts + ((size_t)b * N + j) * 3;
  float* dst = nxyz + (size_t)idx * 3;
  dst[0] = src[0]; dst[1] = src[1]; dst[2] = src[2];
}

// query_ball_point with reference semantics (sorted index order, pad = first)
__global__ void ball_query_kernel(const float* __restrict__ pts,
                                  const float* __restrict__ nxyz,
                                  int* __restrict__ bidx,
                                  int N, int S, int ns, float r2) {
  int idx = blockIdx.x * blockDim.x + threadIdx.x;
  if (idx >= BATCH * S) return;
  int b = idx / S;
  const float cx = nxyz[idx * 3 + 0], cy = nxyz[idx * 3 + 1], cz = nxyz[idx * 3 + 2];
  const float* P = pts + (size_t)b * N * 3;
  int* out = bidx + (size_t)idx * ns;
  int cnt = 0, first = -1;
  for (int j = 0; j < N; ++j) {
    float dx = P[j * 3 + 0] - cx, dy = P[j * 3 + 1] - cy, dz = P[j * 3 + 2] - cz;
    float d = dx * dx + dy * dy + dz * dz;
    if (d <= r2) {
      if (first < 0) first = j;
      if (cnt < ns) out[cnt++] = j;
      if (cnt == ns) break;
    }
  }
  if (first < 0) first = 0;
  for (int k = cnt; k < ns; ++k) out[k] = first;
}

// SA1 grouping: grouped_xyz - center, pad Cin 3 -> 32, f16 output
__global__ void group1_kernel(const float* __restrict__ pts,
                              const float* __restrict__ nxyz,
                              const int* __restrict__ bidx,
                              _Float16* __restrict__ out) {
  int idx = blockIdx.x * blockDim.x + threadIdx.x;   // (b,s,k)
  if (idx >= BATCH * S1 * K1) return;
  int bs = idx / K1;
  int b  = bs / S1;
  int j  = bidx[idx];
  const float* P = pts + ((size_t)b * NPTS + j) * 3;
  const float* C = nxyz + (size_t)bs * 3;
  _Float16* o = out + (size_t)idx * 32;
  o[0] = (_Float16)(P[0] - C[0]);
  o[1] = (_Float16)(P[1] - C[1]);
  o[2] = (_Float16)(P[2] - C[2]);
#pragma unroll
  for (int c = 3; c < 32; ++c) o[c] = (_Float16)0.0f;
}

// SA2 grouping: [xyz-center (3) | l1_pts (128)] pad -> 160, f16 output
__global__ void group2_kernel(const float* __restrict__ l1xyz,
                              const float* __restrict__ l1pts,
                              const float* __restrict__ nxyz2,
                              const int* __restrict__ bidx,
                              _Float16* __restrict__ out) {
  int idx = blockIdx.x * blockDim.x + threadIdx.x;   // (b,s,k)
  if (idx >= BATCH * S2 * K2) return;
  int bs = idx / K2;
  int b  = bs / S2;
  int j  = bidx[idx];
  const float* P = l1xyz + ((size_t)b * S1 + j) * 3;
  const float* F = l1pts + ((size_t)b * S1 + j) * 128;
  const float* C = nxyz2 + (size_t)bs * 3;
  _Float16* o = out + (size_t)idx * 160;
  o[0] = (_Float16)(P[0] - C[0]);
  o[1] = (_Float16)(P[1] - C[1]);
  o[2] = (_Float16)(P[2] - C[2]);
  for (int c = 0; c < 128; ++c) o[3 + c] = (_Float16)F[c];
#pragma unroll
  for (int c = 131; c < 160; ++c) o[c] = (_Float16)0.0f;
}

// SA3 input: [l2_xyz (3) | l2_pts (256)] pad -> 288, rows = B*128
__global__ void concat3_kernel(const float* __restrict__ l2xyz,
                               const float* __restrict__ l2pts,
                               _Float16* __restrict__ out) {
  int idx = blockIdx.x * blockDim.x + threadIdx.x;   // (b, j)
  if (idx >= BATCH * NS3) return;
  const float* X = l2xyz + (size_t)idx * 3;
  const float* F = l2pts + (size_t)idx * 256;
  _Float16* o = out + (size_t)idx * 288;
  o[0] = (_Float16)X[0]; o[1] = (_Float16)X[1]; o[2] = (_Float16)X[2];
  for (int c = 0; c < 256; ++c) o[3 + c] = (_Float16)F[c];
#pragma unroll
  for (int c = 259; c < 288; ++c) o[c] = (_Float16)0.0f;
}

// f32 weights [O,Cin] -> zero-padded f16 [Opad,Cpad]; bias -> padded f32
__global__ void wconv_kernel(const float* __restrict__ w,
                             const float* __restrict__ bias,
                             _Float16* __restrict__ wh,
                             float* __restrict__ bh,
                             int O, int Cin, int Opad, int Cpad) {
  int idx = blockIdx.x * blockDim.x + threadIdx.x;
  int total = Opad * Cpad;
  if (idx < total) {
    int o = idx / Cpad, c = idx % Cpad;
    wh[idx] = (o < O && c < Cin) ? (_Float16)w[(size_t)o * Cin + c] : (_Float16)0.0f;
  }
  if (idx < Opad) bh[idx] = (idx < O) ? bias[idx] : 0.0f;
}

// WMMA GEMM: out[m,o] = sum_c A[m,c] * W[o,c] + bias[o]   (f16 in, f16 out)
// 128-thread blocks = 4 waves; wave w owns M-tile (blockIdx.x*4+w) and the
// four N-tiles of N-strip blockIdx.y (Opad must be a multiple of 64).
// One A fragment feeds 4 independent WMMA chains, straight-line code.
__global__ void wmma_gemm_kernel(const _Float16* __restrict__ A,
                                 const _Float16* __restrict__ W,
                                 const float* __restrict__ bias,
                                 _Float16* __restrict__ out,
                                 int M, int Cpad, int Opad) {
  const int wave = threadIdx.x >> 5;
  const int lane = threadIdx.x & 31;
  const int tm = blockIdx.x * 4 + wave;
  if (tm * 16 >= M) return;                 // wave-uniform: EXEC all-1s below
  const int ln = lane & 15;
  const int hi = lane >> 4;                 // lane group 0 / 1
  const int m = tm * 16 + ln;
  const uint32_t* Arow = reinterpret_cast<const uint32_t*>(A + (size_t)m * Cpad);
  const int nstrip = blockIdx.y * 64;
  const uint32_t* Wrow0 = reinterpret_cast<const uint32_t*>(W + (size_t)(nstrip + 0  + ln) * Cpad);
  const uint32_t* Wrow1 = reinterpret_cast<const uint32_t*>(W + (size_t)(nstrip + 16 + ln) * Cpad);
  const uint32_t* Wrow2 = reinterpret_cast<const uint32_t*>(W + (size_t)(nstrip + 32 + ln) * Cpad);
  const uint32_t* Wrow3 = reinterpret_cast<const uint32_t*>(W + (size_t)(nstrip + 48 + ln) * Cpad);
  v8f acc0 = {}, acc1 = {}, acc2 = {}, acc3 = {};
  for (int k0 = 0; k0 < Cpad; k0 += 32) {
    if (k0 + 64 < Cpad)                      // uniform; speculative prefetch
      __builtin_prefetch(Arow + (k0 >> 1) + 32, 0, 1);
    const int ka = (k0 >> 1) + hi * 8;       // A: contiguous K pairs per lane
    v8u ap;
#pragma unroll
    for (int v = 0; v < 4; ++v) ap[v] = Arow[ka + v];
#pragma unroll
    for (int v = 0; v < 4; ++v) ap[4 + v] = Arow[ka + 4 + v];
    const v16h av = __builtin_bit_cast(v16h, ap);
    const int kb = (k0 >> 1) + hi * 8;       // B: contiguous K pairs per lane
    v8u bp0, bp1, bp2, bp3;
#pragma unroll
    for (int v = 0; v < 8; ++v) bp0[v] = Wrow0[kb + v];
#pragma unroll
    for (int v = 0; v < 8; ++v) bp1[v] = Wrow1[kb + v];
#pragma unroll
    for (int v = 0; v < 8; ++v) bp2[v] = Wrow2[kb + v];
#pragma unroll
    for (int v = 0; v < 8; ++v) bp3[v] = Wrow3[kb + v];
    acc0 = __builtin_amdgcn_wmma_f32_16x16x32_f16(
        false, av, false, __builtin_bit_cast(v16h, bp0), (short)0, acc0, false, false);
    acc1 = __builtin_amdgcn_wmma_f32_16x16x32_f16(
        false, av, false, __builtin_bit_cast(v16h, bp1), (short)0, acc1, false, false);
    acc2 = __builtin_amdgcn_wmma_f32_16x16x32_f16(
        false, av, false, __builtin_bit_cast(v16h, bp2), (short)0, acc2, false, false);
    acc3 = __builtin_amdgcn_wmma_f32_16x16x32_f16(
        false, av, false, __builtin_bit_cast(v16h, bp3), (short)0, acc3, false, false);
  }
  const int mbase = tm * 16 + hi * 8;        // C layout: VGPR r -> row r (+8 for hi)
  const v8f accs[4] = {acc0, acc1, acc2, acc3};
#pragma unroll
  for (int t = 0; t < 4; ++t) {
    const int n = nstrip + t * 16 + ln;
    const float bv = bias[n];
#pragma unroll
    for (int r = 0; r < 8; ++r) {
      out[(size_t)(mbase + r) * Opad + n] = (_Float16)(accs[t][r] + bv);
    }
  }
}

// instance-norm stats: one block per (b, channel); Mb rows per batch
__global__ void in_stats_kernel(const _Float16* __restrict__ raw,
                                float* __restrict__ stats, int Mb, int Opad) {
  __shared__ float s1[256], s2[256];
  const int b = blockIdx.x / Opad, o = blockIdx.x % Opad, tid = threadIdx.x;
  const _Float16* base = raw + (size_t)b * Mb * Opad + o;
  float a = 0.0f, q = 0.0f;
  for (int r = tid; r < Mb; r += 256) {
    float v = (float)base[(size_t)r * Opad];
    a += v; q += v * v;
  }
  s1[tid] = a; s2[tid] = q;
  __syncthreads();
  for (int s = 128; s > 0; s >>= 1) {
    if (tid < s) { s1[tid] += s1[tid + s]; s2[tid] += s2[tid + s]; }
    __syncthreads();
  }
  if (tid == 0) {
    float mean = s1[0] / (float)Mb;
    float var = s2[0] / (float)Mb - mean * mean;
    if (var < 0.0f) var = 0.0f;
    stats[2 * (size_t)blockIdx.x + 0] = mean;
    stats[2 * (size_t)blockIdx.x + 1] = rsqrtf(var + 1e-5f);
  }
}

// in-place normalize + relu on the f16 activation buffer
__global__ void in_norm_relu_kernel(_Float16* __restrict__ raw,
                                    const float* __restrict__ stats,
                                    int Mb, int Opad, int total) {
  int idx = blockIdx.x * blockDim.x + threadIdx.x;
  if (idx >= total) return;
  int o = idx % Opad;
  int b = (idx / Opad) / Mb;
  float mean = stats[2 * ((size_t)b * Opad + o) + 0];
  float inv  = stats[2 * ((size_t)b * Opad + o) + 1];
  float v = ((float)raw[idx] - mean) * inv;
  raw[idx] = (_Float16)(v > 0.0f ? v : 0.0f);
}

// max over the K (nsample) axis -> fp32 [B,S,O]
__global__ void maxpool_kernel(const _Float16* __restrict__ act,
                               float* __restrict__ out,
                               int S, int K, int O) {
  int idx = blockIdx.x * blockDim.x + threadIdx.x;   // (b,s,o)
  if (idx >= BATCH * S * O) return;
  int o = idx % O;
  int bs = idx / O;
  const _Float16* base = act + ((size_t)bs * K) * O + o;
  float m = -1.0e30f;
  for (int k = 0; k < K; ++k) m = fmaxf(m, (float)base[(size_t)k * O]);
  out[idx] = m;
}

// SA3 max over the 128 grouped points -> f16 [B,1024] (feeds fc1 directly)
__global__ void maxpool3_kernel(const _Float16* __restrict__ act,
                                _Float16* __restrict__ out) {
  int idx = blockIdx.x * blockDim.x + threadIdx.x;   // (b,o)
  if (idx >= BATCH * 1024) return;
  int o = idx % 1024, b = idx / 1024;
  const _Float16* base = act + ((size_t)b * NS3) * 1024 + o;
  float m = -1.0e30f;
  for (int k = 0; k < NS3; ++k) m = fmaxf(m, (float)base[(size_t)k * 1024]);
  out[idx] = (_Float16)m;
}

// instnorm1d (per-row over channels) + relu, f16 -> f16
__global__ void instnorm1d_kernel(const _Float16* __restrict__ raw,
                                  _Float16* __restrict__ out, int O) {
  __shared__ float s1[256], s2[256];
  const int b = blockIdx.x, tid = threadIdx.x;
  float a = 0.0f, q = 0.0f;
  for (int c = tid; c < O; c += 256) {
    float v = (float)raw[(size_t)b * O + c];
    a += v; q += v * v;
  }
  s1[tid] = a; s2[tid] = q;
  __syncthreads();
  for (int s = 128; s > 0; s >>= 1) {
    if (tid < s) { s1[tid] += s1[tid + s]; s2[tid] += s2[tid + s]; }
    __syncthreads();
  }
  float mean = s1[0] / (float)O;
  float var = s2[0] / (float)O - mean * mean;
  if (var < 0.0f) var = 0.0f;
  float inv = rsqrtf(var + 1e-5f);
  for (int c = tid; c < O; c += 256) {
    float v = ((float)raw[(size_t)b * O + c] - mean) * inv;
    out[(size_t)b * O + c] = (_Float16)(v > 0.0f ? v : 0.0f);
  }
}

// log_softmax over 12 logits (raw padded to 64 cols) -> fp32 output
__global__ void logsoftmax_kernel(const _Float16* __restrict__ raw,
                                  float* __restrict__ out) {
  int b = blockIdx.x * blockDim.x + threadIdx.x;
  if (b >= BATCH) return;
  float x[12], mx = -1.0e30f;
#pragma unroll
  for (int i = 0; i < 12; ++i) { x[i] = (float)raw[b * 64 + i]; mx = fmaxf(mx, x[i]); }
  float s = 0.0f;
#pragma unroll
  for (int i = 0; i < 12; ++i) s += expf(x[i] - mx);
  float l = logf(s);
#pragma unroll
  for (int i = 0; i < 12; ++i) out[b * 12 + i] = x[i] - mx - l;
}

// ----------------------------- host-side helpers ---------------------------

static inline int cdiv(int a, int b) { return (a + b - 1) / b; }

static void launch_gemm(const _Float16* src, const float* w, const float* b,
                        int O, int Cin, int Opad, int Cpad,
                        _Float16* wh, float* bh, _Float16* dst,
                        int M, hipStream_t stream) {
  // Opad must be a multiple of 64, Cpad a multiple of 32, M a multiple of 16.
  wconv_kernel<<<cdiv(Opad * Cpad, 256), 256, 0, stream>>>(w, b, wh, bh, O, Cin, Opad, Cpad);
  dim3 grid(cdiv(M / 16, 4), Opad / 64);
  wmma_gemm_kernel<<<grid, 128, 0, stream>>>(src, wh, bh, dst, M, Cpad, Opad);
}

static void launch_mlp_layer(const _Float16* src, const float* w, const float* b,
                             int O, int Cin, int Opad, int Cpad,
                             _Float16* wh, float* bh, float* stats,
                             _Float16* dst, int M, int Mb, hipStream_t stream) {
  launch_gemm(src, w, b, O, Cin, Opad, Cpad, wh, bh, dst, M, stream);
  in_stats_kernel<<<BATCH * Opad, 256, 0, stream>>>(dst, stats, Mb, Opad);
  int total = M * Opad;
  in_norm_relu_kernel<<<cdiv(total, 256), 256, 0, stream>>>(dst, stats, Mb, Opad, total);
}

// ----------------------------- entry point ---------------------------------

extern "C" void kernel_launch(void* const* d_in, const int* in_sizes, int n_in,
                              void* d_out, int out_size, void* d_ws, size_t ws_size,
                              hipStream_t stream) {
  (void)in_sizes; (void)n_in; (void)out_size; (void)ws_size;

  // Input order: xyz, then params flattened with dict keys sorted
  // (fc1, fc2, fc3, sa1, sa2, sa3), each layer as (w, b).
  const float* xyz   = (const float*)d_in[0];
  const float* fc1w  = (const float*)d_in[1];
  const float* fc1b  = (const float*)d_in[2];
  const float* fc2w  = (const float*)d_in[3];
  const float* fc2b  = (const float*)d_in[4];
  const float* fc3w  = (const float*)d_in[5];
  const float* fc3b  = (const float*)d_in[6];
  const float* sa1w[3] = {(const float*)d_in[7],  (const float*)d_in[9],  (const float*)d_in[11]};
  const float* sa1b[3] = {(const float*)d_in[8],  (const float*)d_in[10], (const float*)d_in[12]};
  const float* sa2w[3] = {(const float*)d_in[13], (const float*)d_in[15], (const float*)d_in[17]};
  const float* sa2b[3] = {(const float*)d_in[14], (const float*)d_in[16], (const float*)d_in[18]};
  const float* sa3w[3] = {(const float*)d_in[19], (const float*)d_in[21], (const float*)d_in[23]};
  const float* sa3b[3] = {(const float*)d_in[20], (const float*)d_in[22], (const float*)d_in[24]};

  char* ws = (char*)d_ws;
  float*     pts   = (float*)(ws + O_PTS);
  int*       fid1  = (int*)(ws + O_FID1);
  int*       fid2  = (int*)(ws + O_FID2);
  float*     nx1   = (float*)(ws + O_NX1);
  float*     nx2   = (float*)(ws + O_NX2);
  int*       bq1   = (int*)(ws + O_BQ1);
  int*       bq2   = (int*)(ws + O_BQ2);
  float*     l1p   = (float*)(ws + O_L1P);
  float*     l2p   = (float*)(ws + O_L2P);
  float*     stats = (float*)(ws + O_STAT);
  _Float16*  wh    = (_Float16*)(ws + O_WH);
  float*     bh    = (float*)(ws + O_BH);
  _Float16*  bufA  = (_Float16*)(ws + O_BUFA);
  _Float16*  bufB  = (_Float16*)(ws + O_BUFB);
  float*     outp  = (float*)d_out;

  // ---- point cloud prep --------------------------------------------------
  transpose_xyz_kernel<<<cdiv(BATCH * NPTS, 256), 256, 0, stream>>>(xyz, pts);

  // ---- SA1 ---------------------------------------------------------------
  fps_kernel<<<BATCH, 256, 0, stream>>>(pts, fid1, NPTS, S1);
  gather3_kernel<<<cdiv(BATCH * S1, 256), 256, 0, stream>>>(pts, fid1, nx1, NPTS, S1);
  ball_query_kernel<<<cdiv(BATCH * S1, 256), 256, 0, stream>>>(pts, nx1, bq1, NPTS, S1, K1, 0.04f);
  group1_kernel<<<cdiv(BATCH * S1 * K1, 256), 256, 0, stream>>>(pts, nx1, bq1, bufA);

  const int M1 = BATCH * S1 * K1, Mb1 = S1 * K1;
  launch_mlp_layer(bufA, sa1w[0], sa1b[0],  64,  3,  64,  32, wh, bh, stats, bufB, M1, Mb1, stream);
  launch_mlp_layer(bufB, sa1w[1], sa1b[1],  64, 64,  64,  64, wh, bh, stats, bufA, M1, Mb1, stream);
  launch_mlp_layer(bufA, sa1w[2], sa1b[2], 128, 64, 128,  64, wh, bh, stats, bufB, M1, Mb1, stream);
  maxpool_kernel<<<cdiv(BATCH * S1 * 128, 256), 256, 0, stream>>>(bufB, l1p, S1, K1, 128);

  // ---- SA2 ---------------------------------------------------------------
  fps_kernel<<<BATCH, 256, 0, stream>>>(nx1, fid2, S1, S2);
  gather3_kernel<<<cdiv(BATCH * S2, 256), 256, 0, stream>>>(nx1, fid2, nx2, S1, S2);
  ball_query_kernel<<<cdiv(BATCH * S2, 256), 256, 0, stream>>>(nx1, nx2, bq2, S1, S2, K2, 0.16f);
  group2_kernel<<<cdiv(BATCH * S2 * K2, 256), 256, 0, stream>>>(nx1, l1p, nx2, bq2, bufA);

  const int M2 = BATCH * S2 * K2, Mb2 = S2 * K2;
  launch_mlp_layer(bufA, sa2w[0], sa2b[0], 128, 131, 128, 160, wh, bh, stats, bufB, M2, Mb2, stream);
  launch_mlp_layer(bufB, sa2w[1], sa2b[1], 128, 128, 128, 128, wh, bh, stats, bufA, M2, Mb2, stream);
  launch_mlp_layer(bufA, sa2w[2], sa2b[2], 256, 128, 256, 128, wh, bh, stats, bufB, M2, Mb2, stream);
  maxpool_kernel<<<cdiv(BATCH * S2 * 256, 256), 256, 0, stream>>>(bufB, l2p, S2, K2, 256);

  // ---- SA3 (group all) ---------------------------------------------------
  concat3_kernel<<<cdiv(BATCH * NS3, 256), 256, 0, stream>>>(nx2, l2p, bufA);
  const int M3 = BATCH * NS3, Mb3 = NS3;
  launch_mlp_layer(bufA, sa3w[0], sa3b[0],  256, 259,  256, 288, wh, bh, stats, bufB, M3, Mb3, stream);
  launch_mlp_layer(bufB, sa3w[1], sa3b[1],  512, 256,  512, 256, wh, bh, stats, bufA, M3, Mb3, stream);
  launch_mlp_layer(bufA, sa3w[2], sa3b[2], 1024, 512, 1024, 512, wh, bh, stats, bufB, M3, Mb3, stream);
  maxpool3_kernel<<<cdiv(BATCH * 1024, 256), 256, 0, stream>>>(bufB, bufA);  // f16 x [32,1024]

  // ---- FC head -----------------------------------------------------------
  launch_gemm(bufA, fc1w, fc1b, 512, 1024, 512, 1024, wh, bh, bufB, BATCH, stream);
  instnorm1d_kernel<<<BATCH, 256, 0, stream>>>(bufB, bufA, 512);
  launch_gemm(bufA, fc2w, fc2b, 256, 512, 256, 512, wh, bh, bufB, BATCH, stream);
  instnorm1d_kernel<<<BATCH, 256, 0, stream>>>(bufB, bufA, 256);
  launch_gemm(bufA, fc3w, fc3b, 12, 256, 64, 256, wh, bh, bufB, BATCH, stream);
  logsoftmax_kernel<<<1, 32, 0, stream>>>(bufB, outp);
}
